// AutoRegressiveLSTM_87986700026217
// MI455X (gfx1250) — matrix operand order repaired and checked by reading, not compiled, not run on hardware
//
#include <hip/hip_runtime.h>

// Problem dims
#define TSTEPS 512
#define BSZ 256
#define ISZ 256
#define OSZ 256
#define HSZ 512
#define KG  1024            // I + O + H (concat GEMM depth)
#define NBLK 64             // persistent blocks (pairs share an H-tile column)
#define NTHR 256            // 8 waves per block
#define LWS (KG + 8)        // padded LDS row stride (ushort) for gate weights
#define LFS (HSZ + 8)       // padded LDS row stride (ushort) for FC weights
#define PPXA (BSZ * KG)     // one hi/lo plane of the activation matrix
#define PPHR (BSZ * HSZ)    // one hi/lo plane of h_raw

typedef __attribute__((ext_vector_type(16))) __bf16        v16bf;
typedef __attribute__((ext_vector_type(8)))  float         v8f;
typedef __attribute__((ext_vector_type(4)))  unsigned int  u32x4;
typedef __attribute__((ext_vector_type(4)))  float         f32x4;
typedef __attribute__((ext_vector_type(4)))  unsigned short u16x4;

union FragBF { v16bf v; u32x4 q[2]; };

__device__ __forceinline__ unsigned short f2bf(float f) {
  unsigned u = __float_as_uint(f);
  u += 0x7FFFu + ((u >> 16) & 1u);          // round-to-nearest-even
  return (unsigned short)(u >> 16);
}
__device__ __forceinline__ float bf2f(unsigned short h) {
  return __uint_as_float((unsigned)h << 16);
}
__device__ __forceinline__ float sigmoidf_(float x) { return 1.0f / (1.0f + __expf(-x)); }

// Device-wide barrier: monotonic counter, release/acquire at agent scope.
__device__ __forceinline__ void grid_sync(unsigned* cnt, unsigned target) {
  __builtin_amdgcn_fence(__ATOMIC_RELEASE, "agent");
  __syncthreads();
  if (threadIdx.x == 0) {
    __hip_atomic_fetch_add(cnt, 1u, __ATOMIC_RELAXED, __HIP_MEMORY_SCOPE_AGENT);
    while (__hip_atomic_load(cnt, __ATOMIC_RELAXED, __HIP_MEMORY_SCOPE_AGENT) < target)
      __builtin_amdgcn_s_sleep(1);
  }
  __syncthreads();
  __builtin_amdgcn_fence(__ATOMIC_ACQUIRE, "agent");
}

// ---------- init: fp32 -> bf16 hi/lo weight repack + XA[0] (x0|z0|h0) + barrier reset ----------
__global__ void lstm_init(const float* __restrict__ inputs, const float* __restrict__ h0,
                          const float* __restrict__ z0,
                          const float* __restrict__ W_ih, const float* __restrict__ W_hh,
                          const float* __restrict__ W_fc,
                          unsigned short* __restrict__ WgHi, unsigned short* __restrict__ WgLo,
                          unsigned short* __restrict__ WfcHi, unsigned short* __restrict__ WfcLo,
                          unsigned short* __restrict__ XA, unsigned* __restrict__ cnt) {
  int tid = blockIdx.x * blockDim.x + threadIdx.x;
  int stride = gridDim.x * blockDim.x;
  if (tid == 0) *cnt = 0u;
  // W_cat[r][k]: k<512 from W_ih (x|z part), else W_hh (h part)
  for (int i = tid; i < 2048 * KG; i += stride) {
    int r = i >> 10, k = i & 1023;
    float v = (k < 512) ? W_ih[r * 512 + k] : W_hh[r * 512 + (k - 512)];
    unsigned short hi = f2bf(v);
    WgHi[i] = hi;
    WgLo[i] = f2bf(v - bf2f(hi));
  }
  for (int i = tid; i < OSZ * HSZ; i += stride) {
    float v = W_fc[i];
    unsigned short hi = f2bf(v);
    WfcHi[i] = hi;
    WfcLo[i] = f2bf(v - bf2f(hi));
  }
  // XA buffer 0 = [ inputs[0] | z0 | h0 ], hi plane at +0, lo plane at +PPXA
  for (int i = tid; i < PPXA; i += stride) {
    int b = i >> 10, c = i & 1023;
    float v = (c < ISZ) ? inputs[b * ISZ + c]
            : (c < ISZ + OSZ) ? z0[b * OSZ + (c - ISZ)]
                              : h0[b * HSZ + (c - ISZ - OSZ)];
    unsigned short hi = f2bf(v);
    XA[i] = hi;
    XA[PPXA + i] = f2bf(v - bf2f(hi));
  }
}

// ---------- persistent LSTM: 512 steps, 2 grid syncs per step, bf16x3 precision ----------
__global__ void __launch_bounds__(NTHR, 1)
lstm_persistent(const float* __restrict__ inputs,
                const float* __restrict__ c0,
                const float* __restrict__ out_mask,
                const float* __restrict__ h_mask,
                const float* __restrict__ c_mask,
                const float* __restrict__ b_ih,
                const float* __restrict__ b_hh,
                const float* __restrict__ b_fc,
                const unsigned short* __restrict__ WgHi,
                const unsigned short* __restrict__ WgLo,
                const unsigned short* __restrict__ WfcHi,
                const unsigned short* __restrict__ WfcLo,
                unsigned short* __restrict__ XA,   // [2 buf][2 hi/lo][BSZ][KG] bf16
                unsigned short* __restrict__ HR,   // [2 hi/lo][BSZ][HSZ] bf16 (h_raw)
                float* __restrict__ out,
                unsigned* __restrict__ cnt) {
  extern __shared__ unsigned short smem[];
  unsigned short* ldsWh = smem;                         // 64 x LWS : gate W hi
  unsigned short* ldsWl = smem + 64 * LWS;              // 64 x LWS : gate W lo
  unsigned short* ldsFh = smem + 2 * 64 * LWS;          // 16 x LFS : FC W hi
  unsigned short* ldsFl = smem + 2 * 64 * LWS + 16 * LFS;

  const int blk  = blockIdx.x;
  const int j    = blk >> 1;       // H-tile column 0..31 (also FC column for blk<32)
  const int sub  = blk & 1;        // which half of the batch tiles
  const int tid  = threadIdx.x;
  const int w    = tid >> 5;       // wave 0..7
  const int m    = sub * 8 + w;    // batch tile 0..15
  const int lane = tid & 31;
  const int col  = lane & 15;      // N column / A row selector
  const int rh   = lane >> 4;      // lane half
  const int koff = rh * 8;         // A-fragment K sub-chunk offset

  // --- stage weight slices (hi+lo) into LDS once ---
  for (int idx = tid; idx < 64 * (KG / 8); idx += NTHR) {
    int rr = idx / (KG / 8);
    int ck = idx % (KG / 8);
    int g = rr >> 4, n = rr & 15;
    size_t gsrc = (size_t)(g * HSZ + j * 16 + n) * KG + ck * 8;
    *(u32x4*)(ldsWh + rr * LWS + ck * 8) = *(const u32x4*)(WgHi + gsrc);
    *(u32x4*)(ldsWl + rr * LWS + ck * 8) = *(const u32x4*)(WgLo + gsrc);
  }
  if (blk < 32) {
    for (int idx = tid; idx < 16 * (HSZ / 8); idx += NTHR) {
      int n = idx / (HSZ / 8);
      int ck = idx % (HSZ / 8);
      size_t gsrc = (size_t)(j * 16 + n) * HSZ + ck * 8;
      *(u32x4*)(ldsFh + n * LFS + ck * 8) = *(const u32x4*)(WfcHi + gsrc);
      *(u32x4*)(ldsFl + n * LFS + ck * 8) = *(const u32x4*)(WfcLo + gsrc);
    }
  }
  __syncthreads();

  // --- per-wave persistent state: cell tile + masks + biases (C/D layout) ---
  const int hcol = j * 16 + col;
  float creg[8], cmk[8], hmk[8];
#pragma unroll
  for (int v = 0; v < 8; ++v) {
    int row = m * 16 + rh * 8 + v;
    creg[v] = c0[(size_t)row * HSZ + hcol];
    cmk[v]  = c_mask[(size_t)row * HSZ + hcol];
    hmk[v]  = h_mask[(size_t)row * HSZ + hcol];
  }
  float bias0 = b_ih[0 * HSZ + hcol] + b_hh[0 * HSZ + hcol];
  float bias1 = b_ih[1 * HSZ + hcol] + b_hh[1 * HSZ + hcol];
  float bias2 = b_ih[2 * HSZ + hcol] + b_hh[2 * HSZ + hcol];
  float bias3 = b_ih[3 * HSZ + hcol] + b_hh[3 * HSZ + hcol];

  float omk[8], bfc = 0.f;
  if (blk < 32) {
    int ocol = j * 16 + col;
    bfc = b_fc[ocol];
#pragma unroll
    for (int v = 0; v < 8; ++v)
      omk[v] = out_mask[(size_t)(m * 16 + rh * 8 + v) * OSZ + ocol];
  }

  const v8f vzero = {0.f, 0.f, 0.f, 0.f, 0.f, 0.f, 0.f, 0.f};

  for (int t = 0; t < TSTEPS; ++t) {
    const unsigned short* XAcH = XA + (size_t)(t & 1) * (2 * PPXA);
    const unsigned short* XAcL = XAcH + PPXA;
    unsigned short* XAnH = XA + (size_t)((t + 1) & 1) * (2 * PPXA);
    unsigned short* XAnL = XAnH + PPXA;

    // ================= Phase A: gates GEMM (bf16x3) + LSTM cell update =================
    v8f a0 = vzero, a1 = vzero, a2 = vzero, a3 = vzero;
    const unsigned short* arH = XAcH + (size_t)(m * 16 + col) * KG;
    const unsigned short* arL = XAcL + (size_t)(m * 16 + col) * KG;
    for (int kb = 0; kb < KG; kb += 32) {
      FragBF Ah, Al;
      Ah.q[0] = *(const u32x4*)(arH + kb + koff);
      Ah.q[1] = *(const u32x4*)(arH + kb + 16 + koff);
      Al.q[0] = *(const u32x4*)(arL + kb + koff);
      Al.q[1] = *(const u32x4*)(arL + kb + 16 + koff);
      FragBF Bh, Bl;
      const unsigned short *bh, *bl;
#define GATE_MAC(G, ACC)                                                                   \
      bh = ldsWh + ((G) * 16 + col) * LWS + kb + rh * 8;                                   \
      bl = ldsWl + ((G) * 16 + col) * LWS + kb + rh * 8;                                   \
      Bh.q[0] = *(const u32x4*)bh; Bh.q[1] = *(const u32x4*)(bh + 16);                     \
      Bl.q[0] = *(const u32x4*)bl; Bl.q[1] = *(const u32x4*)(bl + 16);                     \
      ACC = __builtin_amdgcn_wmma_f32_16x16x32_bf16(false, Ah.v, false, Bh.v, (short)0, ACC, false, false); \
      ACC = __builtin_amdgcn_wmma_f32_16x16x32_bf16(false, Ah.v, false, Bl.v, (short)0, ACC, false, false); \
      ACC = __builtin_amdgcn_wmma_f32_16x16x32_bf16(false, Al.v, false, Bh.v, (short)0, ACC, false, false);
      GATE_MAC(0, a0)
      GATE_MAC(1, a1)
      GATE_MAC(2, a2)
      GATE_MAC(3, a3)
#undef GATE_MAC
    }
#pragma unroll
    for (int v = 0; v < 8; ++v) {
      int row = m * 16 + rh * 8 + v;
      float ig = sigmoidf_(a0[v] + bias0);
      float fg = sigmoidf_(a1[v] + bias1);
      float gg = tanhf(a2[v] + bias2);
      float og = sigmoidf_(a3[v] + bias3);
      float cn = fg * creg[v] + ig * gg;
      float hn = og * tanhf(cn);            // raw h (pre-mask) feeds the FC
      float hm = hn * hmk[v];               // masked h carried to next step
      creg[v] = cn * cmk[v];                // masked c carried in registers
      unsigned short hhi = f2bf(hn);
      HR[(size_t)row * HSZ + hcol]        = hhi;
      HR[PPHR + (size_t)row * HSZ + hcol] = f2bf(hn - bf2f(hhi));
      unsigned short mhi = f2bf(hm);
      XAnH[(size_t)row * KG + (ISZ + OSZ) + hcol] = mhi;
      XAnL[(size_t)row * KG + (ISZ + OSZ) + hcol] = f2bf(hm - bf2f(mhi));
      if (t == TSTEPS - 1) {
        out[(size_t)TSTEPS * BSZ * OSZ + (size_t)row * HSZ + hcol] = hm;       // final h
        out[(size_t)TSTEPS * BSZ * OSZ + (size_t)BSZ * HSZ
            + (size_t)row * HSZ + hcol] = creg[v];                             // final c
      }
    }
    grid_sync(cnt, (unsigned)(2 * t + 1) * NBLK);

    // ================= Phase B: FC (bf16x3) + tanh + dropout // input staging =================
    if (blk < 32) {
      v8f zacc = vzero;
      const unsigned short* hrH = HR + (size_t)(m * 16 + col) * HSZ;
      const unsigned short* hrL = HR + PPHR + (size_t)(m * 16 + col) * HSZ;
      for (int kb = 0; kb < HSZ; kb += 32) {
        FragBF Ah, Al, Bh, Bl;
        Ah.q[0] = *(const u32x4*)(hrH + kb + koff);
        Ah.q[1] = *(const u32x4*)(hrH + kb + 16 + koff);
        Al.q[0] = *(const u32x4*)(hrL + kb + koff);
        Al.q[1] = *(const u32x4*)(hrL + kb + 16 + koff);
        const unsigned short* bh = ldsFh + col * LFS + kb + rh * 8;
        const unsigned short* bl = ldsFl + col * LFS + kb + rh * 8;
        Bh.q[0] = *(const u32x4*)bh; Bh.q[1] = *(const u32x4*)(bh + 16);
        Bl.q[0] = *(const u32x4*)bl; Bl.q[1] = *(const u32x4*)(bl + 16);
        zacc = __builtin_amdgcn_wmma_f32_16x16x32_bf16(false, Ah.v, false, Bh.v, (short)0, zacc, false, false);
        zacc = __builtin_amdgcn_wmma_f32_16x16x32_bf16(false, Ah.v, false, Bl.v, (short)0, zacc, false, false);
        zacc = __builtin_amdgcn_wmma_f32_16x16x32_bf16(false, Al.v, false, Bh.v, (short)0, zacc, false, false);
      }
      int ocol = j * 16 + col;
#pragma unroll
      for (int v = 0; v < 8; ++v) {
        int row = m * 16 + rh * 8 + v;
        float z = tanhf(zacc[v] + bfc) * omk[v];
        out[(size_t)t * BSZ * OSZ + (size_t)row * OSZ + ocol] = z;
        unsigned short zhi = f2bf(z);
        XAnH[(size_t)row * KG + ISZ + ocol] = zhi;
        XAnL[(size_t)row * KG + ISZ + ocol] = f2bf(z - bf2f(zhi));
      }
    } else if (t + 1 < TSTEPS) {
      // blocks 32..63: stage inputs[t+1] fp32 -> bf16 hi/lo into XA next x-section
      int gtid = (blk - 32) * NTHR + tid;  // 0..8191
      const f32x4* src = (const f32x4*)(inputs + (size_t)(t + 1) * BSZ * ISZ);
#pragma unroll
      for (int it = 0; it < 2; ++it) {
        int i4 = gtid + it * 8192;         // 0..16383 float4 chunks
        f32x4 xv = src[i4];
        int e = i4 * 4;
        int b = e >> 8;
        int c = e & 255;
        u16x4 ph, pl;
#pragma unroll
        for (int q = 0; q < 4; ++q) {
          unsigned short hi = f2bf(xv[q]);
          ph[q] = hi;
          pl[q] = f2bf(xv[q] - bf2f(hi));
        }
        *(u16x4*)(XAnH + (size_t)b * KG + c) = ph;
        *(u16x4*)(XAnL + (size_t)b * KG + c) = pl;
      }
    }
    grid_sync(cnt, (unsigned)(2 * t + 2) * NBLK);
  }
}

extern "C" void kernel_launch(void* const* d_in, const int* in_sizes, int n_in,
                              void* d_out, int out_size, void* d_ws, size_t ws_size,
                              hipStream_t stream) {
  (void)in_sizes; (void)n_in; (void)out_size; (void)ws_size;
  const float* inputs   = (const float*)d_in[0];
  const float* h0       = (const float*)d_in[1];
  const float* c0       = (const float*)d_in[2];
  const float* z0       = (const float*)d_in[3];
  const float* out_mask = (const float*)d_in[4];
  const float* h_mask   = (const float*)d_in[5];
  const float* c_mask   = (const float*)d_in[6];
  const float* W_ih     = (const float*)d_in[7];
  const float* W_hh     = (const float*)d_in[8];
  const float* b_ih     = (const float*)d_in[9];
  const float* b_hh     = (const float*)d_in[10];
  const float* W_fc     = (const float*)d_in[11];
  const float* b_fc     = (const float*)d_in[12];

  char* ws = (char*)d_ws;
  unsigned short* WgHi  = (unsigned short*)ws;                                  // 4 MB
  unsigned short* WgLo  = (unsigned short*)(ws + (size_t)4  * 1024 * 1024);     // 4 MB
  unsigned short* WfcHi = (unsigned short*)(ws + (size_t)8  * 1024 * 1024);     // 256 KB
  unsigned short* WfcLo = (unsigned short*)(ws + (size_t)8  * 1024 * 1024 + 256 * 1024);
  unsigned short* XA    = (unsigned short*)(ws + (size_t)9  * 1024 * 1024);     // 2 MB
  unsigned short* HR    = (unsigned short*)(ws + (size_t)11 * 1024 * 1024);     // 512 KB
  unsigned*       cnt   = (unsigned*)(ws + (size_t)12 * 1024 * 1024);           // barrier counter

  lstm_init<<<2048, 256, 0, stream>>>(inputs, h0, z0, W_ih, W_hh, W_fc,
                                      WgHi, WgLo, WfcHi, WfcLo, XA, cnt);

  size_t smem_bytes = (size_t)(2 * 64 * LWS + 2 * 16 * LFS) * sizeof(unsigned short); // ~290.5 KB
  lstm_persistent<<<NBLK, NTHR, smem_bytes, stream>>>(
      inputs, c0, out_mask, h_mask, c_mask, b_ih, b_hh, b_fc,
      WgHi, WgLo, WfcHi, WfcLo, XA, HR, (float*)d_out, cnt);
}
